// SelfAttention_88596585382183
// MI455X (gfx1250) — matrix operand order repaired and checked by reading, not compiled, not run on hardware
//
#include <hip/hip_runtime.h>

typedef __attribute__((ext_vector_type(16))) __bf16 v16bf;
typedef __attribute__((ext_vector_type(8)))  __bf16 v8bf;
typedef __attribute__((ext_vector_type(2)))  __bf16 v2bf;
typedef __attribute__((ext_vector_type(8)))  float  v8f;

#define B_  16
#define C_  256
#define T_  1024
#define NH_ 4
#define HD_ 64
#define TD_ 256
#define LDK 72   // padded leading dim (elements) for 64-wide K-contiguous tiles
#define LDK2 72  // K tile rows (64 d elems + pad)
#define LDV 40   // padded leading dim for 32-wide K-contiguous tiles

// ---------------------------------------------------------------- WMMA core
__device__ __forceinline__ v8f wmma_bf16(v16bf a, v16bf b, v8f c) {
  return __builtin_amdgcn_wmma_f32_16x16x32_bf16(false, a, false, b,
                                                 (short)0, c, false, false);
}

__device__ __forceinline__ v16bf cat8(v8bf lo, v8bf hi) {
  return __builtin_shufflevector(lo, hi, 0, 1, 2, 3, 4, 5, 6, 7,
                                 8, 9, 10, 11, 12, 13, 14, 15);
}

// A-frag: 16x32 bf16 tile, row-major (K contiguous), leading dim lda.
__device__ __forceinline__ v16bf load_a_frag(const __bf16* t, int lda) {
  int lane = threadIdx.x & 31, m = lane & 15, half = lane >> 4;
  const __bf16* p = t + m * lda + half * 8;
  return cat8(*(const v8bf*)p, *(const v8bf*)(p + 16));
}

// B-frag from an N-major, K-contiguous tile (t points at [n0][k0], ldk elems).
__device__ __forceinline__ v16bf load_bk_frag(const __bf16* t, int ldk) {
  int lane = threadIdx.x & 31, n = lane & 15, half = lane >> 4;
  const __bf16* p = t + n * ldk + half * 8;
  return cat8(*(const v8bf*)p, *(const v8bf*)(p + 16));
}

// Async global->LDS 16B copy (GLOBAL_LOAD_ASYNC_TO_LDS_B128, ASYNCcnt).
// LDS offset = low 32 bits of the generic pointer (LDS aperture mapping).
__device__ __forceinline__ void async_b128(void* lds, const void* g) {
  unsigned l = (unsigned)(uintptr_t)lds;
  unsigned long long ga = (unsigned long long)(uintptr_t)g;
  asm volatile("global_load_async_to_lds_b128 %0, %1, off"
               :: "v"(l), "v"(ga) : "memory");
}
__device__ __forceinline__ void wait_async0() {
  asm volatile("s_wait_asynccnt 0" ::: "memory");
}

// ------------------------------------------------- 1) te = t_emb @ Wt + bt
__global__ __launch_bounds__(256) void temb_kernel(const float* __restrict__ t_emb,
                                                   const float* __restrict__ Wt,
                                                   const float* __restrict__ bt,
                                                   float* __restrict__ te) {
  int idx = blockIdx.x * 256 + threadIdx.x;  // B*C = 4096 outputs
  int b = idx >> 8, c = idx & 255;
  float acc = bt[c];
  for (int k = 0; k < TD_; ++k)
    acc += t_emb[b * TD_ + k] * Wt[k * C_ + c];
  te[idx] = acc;
}

// --------- 2) qkv = (x^T + te) @ Wqkv + bqkv, scattered to head-major Q/K/V
__global__ __launch_bounds__(256) void qkv_kernel(const float* __restrict__ x,
                                                  const float* __restrict__ te,
                                                  const float* __restrict__ Wqkv,
                                                  const float* __restrict__ bqkv,
                                                  __bf16* __restrict__ Qw,
                                                  __bf16* __restrict__ Kw,
                                                  __bf16* __restrict__ Vw) {
  __shared__ __attribute__((aligned(16))) __bf16 Xs[128][LDK];   // [M][K]
  __shared__ __attribute__((aligned(16))) __bf16 WsT[64][LDK];   // [N][K]
  int tid  = threadIdx.x;
  int wave = tid >> 5, lane = tid & 31, n = lane & 15, half = lane >> 4;
  int row0 = blockIdx.x * 128;  // rows of X' = [B*T, C]
  int n0   = blockIdx.y * 64;   // cols of Wqkv (768 wide)
  int b    = row0 >> 10;        // 128 | 1024 -> one batch per tile
  v8f acc[4] = {};
  for (int k0 = 0; k0 < C_; k0 += 64) {
    // X' tile: 2 K-rows per chunk -> packed bf16x2 stores (ds_store_b32)
    for (int e = tid; e < (128 * 64) / 8; e += 256) {
      int r = (e & 31) * 4, k2 = (e >> 5) * 2;
      int c = k0 + k2;
      const float* xp = x + ((size_t)b * C_ + c) * T_ + row0 + r;
      float4 x0 = *(const float4*)xp;
      float4 x1 = *(const float4*)(xp + T_);
      float t0 = te[b * C_ + c], t1 = te[b * C_ + c + 1];
      v2bf p;
      p[0] = (__bf16)(x0.x + t0); p[1] = (__bf16)(x1.x + t1); *(v2bf*)&Xs[r + 0][k2] = p;
      p[0] = (__bf16)(x0.y + t0); p[1] = (__bf16)(x1.y + t1); *(v2bf*)&Xs[r + 1][k2] = p;
      p[0] = (__bf16)(x0.z + t0); p[1] = (__bf16)(x1.z + t1); *(v2bf*)&Xs[r + 2][k2] = p;
      p[0] = (__bf16)(x0.w + t0); p[1] = (__bf16)(x1.w + t1); *(v2bf*)&Xs[r + 3][k2] = p;
    }
    // W tile transposed to [N][K], 2 K-rows per chunk -> packed b32 stores
    for (int e = tid; e < (64 * 64) / 8; e += 256) {
      int nn = (e & 15) * 4, k2 = (e >> 4) * 2;
      const float* wp = Wqkv + (size_t)(k0 + k2) * 768 + n0 + nn;
      float4 w0 = *(const float4*)wp;
      float4 w1 = *(const float4*)(wp + 768);
      v2bf p;
      p[0] = (__bf16)w0.x; p[1] = (__bf16)w1.x; *(v2bf*)&WsT[nn + 0][k2] = p;
      p[0] = (__bf16)w0.y; p[1] = (__bf16)w1.y; *(v2bf*)&WsT[nn + 1][k2] = p;
      p[0] = (__bf16)w0.z; p[1] = (__bf16)w1.z; *(v2bf*)&WsT[nn + 2][k2] = p;
      p[0] = (__bf16)w0.w; p[1] = (__bf16)w1.w; *(v2bf*)&WsT[nn + 3][k2] = p;
    }
    __syncthreads();
    v16bf a0 = load_a_frag(&Xs[wave * 16][0], LDK);
    v16bf a1 = load_a_frag(&Xs[wave * 16][32], LDK);
#pragma unroll
    for (int nt = 0; nt < 4; ++nt) {
      acc[nt] = wmma_bf16(a0, load_bk_frag(&WsT[nt * 16][0], LDK), acc[nt]);
      acc[nt] = wmma_bf16(a1, load_bk_frag(&WsT[nt * 16][32], LDK), acc[nt]);
    }
    __syncthreads();
  }
  // scatter into [B, h, T, hd] bf16 buffers (fused split_heads)
#pragma unroll
  for (int nt = 0; nt < 4; ++nt)
#pragma unroll
    for (int i = 0; i < 8; ++i) {
      int gm  = row0 + wave * 16 + i + 8 * half;
      int t   = gm & 1023;
      int col = n0 + nt * 16 + n;  // [0,768)
      float v = acc[nt][i] + bqkv[col];
      int sec = col >> 8, c = col & 255, head = c >> 6, d = c & 63;
      __bf16* dst = sec == 0 ? Qw : (sec == 1 ? Kw : Vw);
      dst[(((size_t)b * NH_ + head) * T_ + t) * HD_ + d] = (__bf16)v;
    }
}

// ------------------------------- 3) flash attention per (b, h, 128 q-rows)
__global__ __launch_bounds__(256) void attn_kernel(const __bf16* __restrict__ Qw,
                                                   const __bf16* __restrict__ Kw,
                                                   const __bf16* __restrict__ Vw,
                                                   __bf16* __restrict__ AO) {
  __shared__ __attribute__((aligned(16))) __bf16 Ks[32][LDK2];    // [kcol][d]
  __shared__ __attribute__((aligned(16))) __bf16 VsT[HD_][LDV];   // [d][kcol]
  __shared__ __attribute__((aligned(16))) __bf16 Ps[8][16][LDV];  // per-wave P
  int tid  = threadIdx.x;
  int wave = tid >> 5, lane = tid & 31, n = lane & 15, half = lane >> 4;
  int q0 = blockIdx.x * 128;
  int h  = blockIdx.y, b = blockIdx.z;
  size_t bh = ((size_t)b * NH_ + h) * T_;

  const __bf16* qbase = Qw + (bh + q0 + wave * 16) * HD_;
  v16bf qa0 = load_a_frag(qbase, HD_);       // d = 0..31
  v16bf qa1 = load_a_frag(qbase + 32, HD_);  // d = 32..63

  float mrow[8], lrow[8];
  v8f o[4] = {};
#pragma unroll
  for (int i = 0; i < 8; ++i) { mrow[i] = -1e30f; lrow[i] = 0.f; }

  int sr = tid >> 3;       // staging row 0..31
  int sd = (tid & 7) * 8;  // staging d chunk
  void* ldsK = &Ks[sr][sd];

  for (int kt = 0; kt < T_ / 32; ++kt) {
    size_t g = (bh + kt * 32 + sr) * HD_ + sd;
    // K tile: async DMA straight into LDS (no VGPR round-trip)
    async_b128(ldsK, Kw + g);
    // V tile: transposed on the way in (needs the VGPR hop)
    v8bf vv = *(const v8bf*)(Vw + g);
#pragma unroll
    for (int q = 0; q < 8; ++q) VsT[sd + q][sr] = vv[q];
    if (kt + 1 < T_ / 32) {  // prefetch next tile (global_prefetch_b8)
      size_t gn = g + 32 * HD_;
      __builtin_prefetch(Kw + gn, 0, 1);
      __builtin_prefetch(Vw + gn, 0, 1);
    }
    wait_async0();
    __syncthreads();

    // S tile 16x32 = two 16x16 frags, chained over d (K^T is K-contiguous)
    v8f s0 = {}, s1 = {};
    s0 = wmma_bf16(qa0, load_bk_frag(&Ks[0][0], LDK2), s0);
    s0 = wmma_bf16(qa1, load_bk_frag(&Ks[0][32], LDK2), s0);
    s1 = wmma_bf16(qa0, load_bk_frag(&Ks[16][0], LDK2), s1);
    s1 = wmma_bf16(qa1, load_bk_frag(&Ks[16][32], LDK2), s1);

    const float scale = 0.125f;  // 1/sqrt(64)
#pragma unroll
    for (int i = 0; i < 8; ++i) {
      float a0 = s0[i] * scale, a1 = s1[i] * scale;
      float tm = fmaxf(a0, a1);
#pragma unroll
      for (int off = 1; off < 16; off <<= 1)  // a row spans 16 lanes
        tm = fmaxf(tm, __shfl_xor(tm, off, 32));
      float mnew  = fmaxf(mrow[i], tm);
      float alpha = __expf(mrow[i] - mnew);
      mrow[i] = mnew;
      float p0 = __expf(a0 - mnew), p1 = __expf(a1 - mnew);
      float rs = p0 + p1;
#pragma unroll
      for (int off = 1; off < 16; off <<= 1)
        rs += __shfl_xor(rs, off, 32);
      lrow[i] = lrow[i] * alpha + rs;
#pragma unroll
      for (int nt = 0; nt < 4; ++nt) o[nt][i] *= alpha;
      Ps[wave][i + 8 * half][n]      = (__bf16)p0;
      Ps[wave][i + 8 * half][16 + n] = (__bf16)p1;
    }
    // wave-local P round-trip: same-wave DS ops are in-order; drain DScnt
    asm volatile("s_wait_dscnt 0" ::: "memory");

    v16bf pa = load_a_frag(&Ps[wave][0][0], LDV);  // P in A layout
#pragma unroll
    for (int nt = 0; nt < 4; ++nt)
      o[nt] = wmma_bf16(pa, load_bk_frag(&VsT[nt * 16][0], LDV), o[nt]);
    __syncthreads();
  }

  // normalize + write attn-out as [B, T, C] bf16 (fused merge_heads)
#pragma unroll
  for (int nt = 0; nt < 4; ++nt)
#pragma unroll
    for (int i = 0; i < 8; ++i) {
      int q = q0 + wave * 16 + i + 8 * half;
      float v = o[nt][i] / lrow[i];
      AO[((size_t)b * T_ + q) * C_ + h * HD_ + nt * 16 + n] = (__bf16)v;
    }
}

// ------------- 4) out = AO @ Wproj + bproj, transposed back to [B,C,H,W]
__global__ __launch_bounds__(256) void proj_kernel(const __bf16* __restrict__ AO,
                                                   const float* __restrict__ Wproj,
                                                   const float* __restrict__ bproj,
                                                   float* __restrict__ out) {
  __shared__ __attribute__((aligned(16))) __bf16 Xs[128][LDK];
  __shared__ __attribute__((aligned(16))) __bf16 WsT[64][LDK];
  int tid  = threadIdx.x;
  int wave = tid >> 5, lane = tid & 31, n = lane & 15, half = lane >> 4;
  int row0 = blockIdx.x * 128;
  int n0   = blockIdx.y * 64;
  v8f acc[4] = {};
  for (int k0 = 0; k0 < C_; k0 += 64) {
    // AO is bf16 row-major [M][K]: pure layout-preserving copy -> async DMA
    for (int e = tid; e < (128 * 64) / 8; e += 256) {
      int kk = (e & 7) * 8, r = e >> 3;
      async_b128(&Xs[r][kk], AO + (size_t)(row0 + r) * C_ + k0 + kk);
    }
    for (int e = tid; e < (64 * 64) / 8; e += 256) {
      int nn = (e & 15) * 4, k2 = (e >> 4) * 2;
      const float* wp = Wproj + (size_t)(k0 + k2) * C_ + n0 + nn;
      float4 w0 = *(const float4*)wp;
      float4 w1 = *(const float4*)(wp + C_);
      v2bf p;
      p[0] = (__bf16)w0.x; p[1] = (__bf16)w1.x; *(v2bf*)&WsT[nn + 0][k2] = p;
      p[0] = (__bf16)w0.y; p[1] = (__bf16)w1.y; *(v2bf*)&WsT[nn + 1][k2] = p;
      p[0] = (__bf16)w0.z; p[1] = (__bf16)w1.z; *(v2bf*)&WsT[nn + 2][k2] = p;
      p[0] = (__bf16)w0.w; p[1] = (__bf16)w1.w; *(v2bf*)&WsT[nn + 3][k2] = p;
    }
    wait_async0();
    __syncthreads();
    v16bf a0 = load_a_frag(&Xs[wave * 16][0], LDK);
    v16bf a1 = load_a_frag(&Xs[wave * 16][32], LDK);
#pragma unroll
    for (int nt = 0; nt < 4; ++nt) {
      acc[nt] = wmma_bf16(a0, load_bk_frag(&WsT[nt * 16][0], LDK), acc[nt]);
      acc[nt] = wmma_bf16(a1, load_bk_frag(&WsT[nt * 16][32], LDK), acc[nt]);
    }
    __syncthreads();
  }
#pragma unroll
  for (int nt = 0; nt < 4; ++nt)
#pragma unroll
    for (int i = 0; i < 8; ++i) {
      int gm = row0 + wave * 16 + i + 8 * half;
      int b = gm >> 10, t = gm & 1023;
      int col = n0 + nt * 16 + n;
      out[((size_t)b * C_ + col) * T_ + t] = acc[nt][i] + bproj[col];
    }
}

// --------------------------------------------------------------- launcher
extern "C" void kernel_launch(void* const* d_in, const int* in_sizes, int n_in,
                              void* d_out, int out_size, void* d_ws, size_t ws_size,
                              hipStream_t stream) {
  (void)in_sizes; (void)n_in; (void)out_size; (void)ws_size;
  const float* x     = (const float*)d_in[0];
  const float* t_emb = (const float*)d_in[1];
  const float* Wqkv  = (const float*)d_in[2];
  const float* bqkv  = (const float*)d_in[3];
  const float* Wproj = (const float*)d_in[4];
  const float* bproj = (const float*)d_in[5];
  const float* Wt    = (const float*)d_in[6];
  const float* bt    = (const float*)d_in[7];

  char* ws = (char*)d_ws;
  const size_t nqkv = (size_t)B_ * T_ * C_;  // 4M elems
  float*  te = (float*)ws;          size_t off = (size_t)B_ * C_ * sizeof(float);
  __bf16* Qw = (__bf16*)(ws + off); off += nqkv * sizeof(__bf16);
  __bf16* Kw = (__bf16*)(ws + off); off += nqkv * sizeof(__bf16);
  __bf16* Vw = (__bf16*)(ws + off); off += nqkv * sizeof(__bf16);
  __bf16* AO = (__bf16*)(ws + off);  // total ~32 MB of workspace

  temb_kernel<<<dim3(B_ * C_ / 256), 256, 0, stream>>>(t_emb, Wt, bt, te);
  qkv_kernel<<<dim3((B_ * T_) / 128, 768 / 64), 256, 0, stream>>>(
      x, te, Wqkv, bqkv, Qw, Kw, Vw);
  attn_kernel<<<dim3(T_ / 128, NH_, B_), 256, 0, stream>>>(Qw, Kw, Vw, AO);
  proj_kernel<<<dim3((B_ * T_) / 128, C_ / 64), 256, 0, stream>>>(
      AO, Wproj, bproj, (float*)d_out);
}
